// Attention_84945863180853
// MI455X (gfx1250) — matrix operand order repaired and checked by reading, not compiled, not run on hardware
//
#include <hip/hip_runtime.h>

typedef __attribute__((ext_vector_type(16))) _Float16 v16h;
typedef __attribute__((ext_vector_type(8)))  _Float16 v8h;
typedef __attribute__((ext_vector_type(4)))  _Float16 v4h;
typedef __attribute__((ext_vector_type(8)))  float    v8f;
typedef unsigned int v4u __attribute__((ext_vector_type(4)));
typedef int          v8i __attribute__((ext_vector_type(8)));
typedef int          v4i __attribute__((ext_vector_type(4)));

#define S_LEN 2048
#define DHEAD 64
#define NHEAD 32   // B*H
#define BLK_M 64   // query rows per workgroup (4 waves x 16)
#define BLK_N 64   // keys per outer iteration
#define KPAD  8    // LDS row padding in halfs (row pitch 144 B, 16B-aligned)

#define WMMA_F16(A, B, C) \
  __builtin_amdgcn_wmma_f32_16x16x32_f16(false, (A), false, (B), (short)0, (C), false, false)

#define EXP2(x) __builtin_amdgcn_exp2f(x)

__device__ __forceinline__ v16h cat8(v8h lo, v8h hi) {
  return __builtin_shufflevector(lo, hi, 0, 1, 2, 3, 4, 5, 6, 7, 8, 9, 10, 11, 12, 13, 14, 15);
}

// ---- TDM: 2-D f16 tile (tileD1 rows x tileD0 elems) global -> LDS, with LDS
// row padding of 4 DWORDs after every 32 DWORDs (128B row -> 144B pitch).
__device__ __forceinline__ void tdm_load_2d(unsigned ldsAddr, const _Float16* gptr,
                                            unsigned tensorD0, unsigned tensorD1,
                                            unsigned tileD0, unsigned tileD1,
                                            unsigned stride0) {
  const unsigned long long ga = (unsigned long long)(__SIZE_TYPE__)gptr;
  v4u g0;
  g0[0] = 1u;                                                  // count=1, user D#
  g0[1] = ldsAddr;                                             // LDS byte address
  g0[2] = (unsigned)ga;                                        // global_addr[31:0]
  g0[3] = (unsigned)((ga >> 32) & 0x01FFFFFFu) | (2u << 30);   // addr[56:32] | type=2
  v8i g1;
  // data_size=1 (2B) | pad_enable | pad_interval=4 (32 dwords) | pad_amount=3 (4 dwords)
  g1[0] = (int)((1u << 16) | (1u << 20) | (4u << 22) | (3u << 25));
  g1[1] = (int)((tensorD0 & 0xFFFFu) << 16);                   // tensor_dim0[15:0]
  g1[2] = (int)(((tensorD0 >> 16) & 0xFFFFu) | ((tensorD1 & 0xFFFFu) << 16));
  g1[3] = (int)(((tensorD1 >> 16) & 0xFFFFu) | (tileD0 << 16)); // tile_dim0
  g1[4] = (int)(tileD1 & 0xFFFFu);                              // tile_dim1, tile_dim2=0
  g1[5] = (int)stride0;                                         // tensor_dim0_stride[31:0]
  g1[6] = 0;
  g1[7] = 0;
  const v4i z4 = {0, 0, 0, 0};
#if defined(__clang_major__) && __clang_major__ >= 23
  const v8i z8 = {0, 0, 0, 0, 0, 0, 0, 0};
  __builtin_amdgcn_tensor_load_to_lds(g0, g1, z4, z4, z8, 0);
#else
  __builtin_amdgcn_tensor_load_to_lds(g0, g1, z4, z4, 0);
#endif
}

// ---- pre-pass 1: K16 = (float)K * (1/sqrt(D) * log2 e) as f16, same layout
__global__ __launch_bounds__(256) void conv_k(const float* __restrict__ K,
                                              _Float16* __restrict__ K16) {
  const float cs = 0.125f * 1.44269504088896f;
  const size_t i = ((size_t)blockIdx.x * 256 + threadIdx.x) * 4;
  const float4 f = *(const float4*)(K + i);
  v4h h;
  h[0] = (_Float16)(f.x * cs); h[1] = (_Float16)(f.y * cs);
  h[2] = (_Float16)(f.z * cs); h[3] = (_Float16)(f.w * cs);
  *(v4h*)(K16 + i) = h;
}

// ---- pre-pass 2: V16t[bh][d][s] = (f16) V[bh][s][d]  (4x4 register transpose)
__global__ __launch_bounds__(256) void conv_v(const float* __restrict__ V,
                                              _Float16* __restrict__ V16t) {
  const int t  = blockIdx.x * 256 + threadIdx.x;   // 0 .. 32*512*16-1
  const int dq = t & 15;                           // d block   (16 x 4)
  const int kq = (t >> 4) & 511;                   // key block (512 x 4)
  const int bh = t >> 13;
  const float* src = V + ((size_t)bh * S_LEN + kq * 4) * DHEAD + dq * 4;
  float vv[4][4];
#pragma unroll
  for (int j = 0; j < 4; ++j) {
    const float4 f = *(const float4*)(src + (size_t)j * DHEAD);
    vv[j][0] = f.x; vv[j][1] = f.y; vv[j][2] = f.z; vv[j][3] = f.w;
  }
  _Float16* dst = V16t + ((size_t)bh * DHEAD + dq * 4) * S_LEN + kq * 4;
#pragma unroll
  for (int i = 0; i < 4; ++i) {
    v4h h;
    h[0] = (_Float16)vv[0][i]; h[1] = (_Float16)vv[1][i];
    h[2] = (_Float16)vv[2][i]; h[3] = (_Float16)vv[3][i];
    *(v4h*)(dst + (size_t)i * S_LEN) = h;
  }
}

// ---- main: transposed-score flash attention, TDM double-buffered tiles ----
// S^T = K Q^T, O^T = V^T P^T. Query index on lanes (1 row/lane), key index on
// VGPRs -> in-lane softmax reductions + one xor-16 shuffle.
__global__ __launch_bounds__(128) void fa_fwd_causal(const float* __restrict__ Q,
                                                     const _Float16* __restrict__ K16,
                                                     const _Float16* __restrict__ V16t,
                                                     float* __restrict__ O) {
  __shared__ __align__(16) _Float16 Ks[2][BLK_N][DHEAD + KPAD];   // [buf][key][d]
  __shared__ __align__(16) _Float16 Vt[2][DHEAD][BLK_N + KPAD];   // [buf][d][key]
  __shared__ __align__(16) _Float16 Pst[4][16][BLK_N + KPAD];     // [wave][m][key]

  const int tid  = threadIdx.x;
  const int lane = tid & 31;
  const int w    = tid >> 5;
  const int g    = lane >> 4;
  const int cN   = lane & 15;
  const int qb   = blockIdx.x;
  const int bh   = blockIdx.y;
  const int mBase = qb * BLK_M;
  const int rowW  = mBase + w * 16;
  const int rowG  = rowW + cN;
  const int wu    = __builtin_amdgcn_readfirstlane(w);

  const float*    Qh = Q    + (size_t)bh * S_LEN * DHEAD;
  const _Float16* Kh = K16  + (size_t)bh * S_LEN * DHEAD;   // [key][d]
  const _Float16* Vh = V16t + (size_t)bh * DHEAD * S_LEN;   // [d][key]
  float*          Oh = O    + (size_t)bh * S_LEN * DHEAD;

  // ---- Q^T as WMMA B-operand (scale already folded into K16) ----
  v16h qB[2];
  {
    const float* qrow = Qh + (size_t)rowG * DHEAD;
#pragma unroll
    for (int c2 = 0; c2 < 2; ++c2) {
      const float4 f0 = *(const float4*)(qrow + 32 * c2 + 8 * g);
      const float4 f1 = *(const float4*)(qrow + 32 * c2 + 8 * g + 4);
      const float4 f2 = *(const float4*)(qrow + 32 * c2 + 16 + 8 * g);
      const float4 f3 = *(const float4*)(qrow + 32 * c2 + 16 + 8 * g + 4);
      v16h t;
      t[0]  = (_Float16)f0.x; t[1]  = (_Float16)f0.y;
      t[2]  = (_Float16)f0.z; t[3]  = (_Float16)f0.w;
      t[4]  = (_Float16)f1.x; t[5]  = (_Float16)f1.y;
      t[6]  = (_Float16)f1.z; t[7]  = (_Float16)f1.w;
      t[8]  = (_Float16)f2.x; t[9]  = (_Float16)f2.y;
      t[10] = (_Float16)f2.z; t[11] = (_Float16)f2.w;
      t[12] = (_Float16)f3.x; t[13] = (_Float16)f3.y;
      t[14] = (_Float16)f3.z; t[15] = (_Float16)f3.w;
      qB[c2] = t;
    }
  }

  v8f acc0 = {}, acc1 = {}, acc2 = {}, acc3 = {};
  float mrow = -__builtin_inff();
  float lrow = 0.0f;

  // ---- preload buffer 0 via TDM (wave 0 issues; barrier publishes) ----
  if (wu == 0) {
    tdm_load_2d((unsigned)(__SIZE_TYPE__)&Ks[0][0][0], Kh,
                DHEAD, S_LEN, DHEAD, BLK_N, DHEAD);
    tdm_load_2d((unsigned)(__SIZE_TYPE__)&Vt[0][0][0], Vh,
                S_LEN, DHEAD, BLK_N, DHEAD, S_LEN);
    __builtin_amdgcn_s_wait_tensorcnt(0);
  }
  __syncthreads();

  for (int kb = 0; kb <= qb; ++kb) {
    const int cur = kb & 1;
    // issue DMA for the next tile into the other buffer; overlaps compute
    if (wu == 0 && kb < qb) {
      tdm_load_2d((unsigned)(__SIZE_TYPE__)&Ks[cur ^ 1][0][0],
                  Kh + (size_t)(kb + 1) * BLK_N * DHEAD,
                  DHEAD, S_LEN, DHEAD, BLK_N, DHEAD);
      tdm_load_2d((unsigned)(__SIZE_TYPE__)&Vt[cur ^ 1][0][0],
                  Vh + (size_t)(kb + 1) * BLK_N,
                  S_LEN, DHEAD, BLK_N, DHEAD, S_LEN);
    }

    const bool diag = (kb == qb);
    const bool full = (kb < qb) || (wu >= 2);   // scalar: keys 32..63 needed?

    // ---- S^T tiles: st_t = K[t*16:(t+1)*16] Q^T ----
    v8f st0 = {}, st1 = {}, st2 = {}, st3 = {};
#pragma unroll
    for (int c2 = 0; c2 < 2; ++c2) {
      const v16h a0 = cat8(*(const v8h*)&Ks[cur][cN][32 * c2 + 8 * g],
                           *(const v8h*)&Ks[cur][cN][32 * c2 + 16 + 8 * g]);
      const v16h a1 = cat8(*(const v8h*)&Ks[cur][16 + cN][32 * c2 + 8 * g],
                           *(const v8h*)&Ks[cur][16 + cN][32 * c2 + 16 + 8 * g]);
      st0 = WMMA_F16(a0, qB[c2], st0);
      st1 = WMMA_F16(a1, qB[c2], st1);
    }
    if (full) {
#pragma unroll
      for (int c2 = 0; c2 < 2; ++c2) {
        const v16h a2 = cat8(*(const v8h*)&Ks[cur][32 + cN][32 * c2 + 8 * g],
                             *(const v8h*)&Ks[cur][32 + cN][32 * c2 + 16 + 8 * g]);
        const v16h a3 = cat8(*(const v8h*)&Ks[cur][48 + cN][32 * c2 + 8 * g],
                             *(const v8h*)&Ks[cur][48 + cN][32 * c2 + 16 + 8 * g]);
        st2 = WMMA_F16(a2, qB[c2], st2);
        st3 = WMMA_F16(a3, qB[c2], st3);
      }
    }

    // ---- causal mask ----
    float x0[8], x1[8], x2[8], x3[8];
    const int keyB = kb * BLK_N + 8 * g;
    if (diag) {
#pragma unroll
      for (int v = 0; v < 8; ++v) {
        x0[v] = (keyB + v      <= rowG) ? st0[v] : -__builtin_inff();
        x1[v] = (keyB + 16 + v <= rowG) ? st1[v] : -__builtin_inff();
      }
    } else {
#pragma unroll
      for (int v = 0; v < 8; ++v) { x0[v] = st0[v]; x1[v] = st1[v]; }
    }
    if (full) {
      if (diag) {
#pragma unroll
        for (int v = 0; v < 8; ++v) {
          x2[v] = (keyB + 32 + v <= rowG) ? st2[v] : -__builtin_inff();
          x3[v] = (keyB + 48 + v <= rowG) ? st3[v] : -__builtin_inff();
        }
      } else {
#pragma unroll
        for (int v = 0; v < 8; ++v) { x2[v] = st2[v]; x3[v] = st3[v]; }
      }
    }

    // ---- online softmax ----
    float tmax = fmaxf(x0[0], x1[0]);
#pragma unroll
    for (int v = 1; v < 8; ++v) tmax = fmaxf(tmax, fmaxf(x0[v], x1[v]));
    if (full) {
#pragma unroll
      for (int v = 0; v < 8; ++v) tmax = fmaxf(tmax, fmaxf(x2[v], x3[v]));
    }
    tmax = fmaxf(tmax, __shfl_xor(tmax, 16));
    const float mn   = fmaxf(mrow, tmax);
    const float corr = EXP2(mrow - mn);
    v8h ph0, ph1, ph2, ph3;
    float rs = 0.0f;
#pragma unroll
    for (int v = 0; v < 8; ++v) {
      const float p0 = EXP2(x0[v] - mn);
      const float p1 = EXP2(x1[v] - mn);
      rs += p0 + p1;
      ph0[v] = (_Float16)p0;
      ph1[v] = (_Float16)p1;
    }
    if (full) {
#pragma unroll
      for (int v = 0; v < 8; ++v) {
        const float p2 = EXP2(x2[v] - mn);
        const float p3 = EXP2(x3[v] - mn);
        rs += p2 + p3;
        ph2[v] = (_Float16)p2;
        ph3[v] = (_Float16)p3;
      }
    }
    rs += __shfl_xor(rs, 16);
    lrow = lrow * corr + rs;
    mrow = mn;
#pragma unroll
    for (int v = 0; v < 8; ++v) {
      acc0[v] *= corr; acc1[v] *= corr; acc2[v] *= corr; acc3[v] *= corr;
    }

    // ---- P^T staging (aligned b128 LDS ops; per-wave slab) ----
    *(v8h*)&Pst[w][cN][8 * g]      = ph0;
    *(v8h*)&Pst[w][cN][16 + 8 * g] = ph1;
    if (full) {
      *(v8h*)&Pst[w][cN][32 + 8 * g] = ph2;
      *(v8h*)&Pst[w][cN][48 + 8 * g] = ph3;
    }
    const v16h pB0 = cat8(*(const v8h*)&Pst[w][cN][8 * g],
                          *(const v8h*)&Pst[w][cN][16 + 8 * g]);

    // ---- O^T += V^T P^T : keys 0..31 ----
    {
      const v16h av0 = cat8(*(const v8h*)&Vt[cur][cN][8 * g],
                            *(const v8h*)&Vt[cur][cN][16 + 8 * g]);
      acc0 = WMMA_F16(av0, pB0, acc0);
      const v16h av1 = cat8(*(const v8h*)&Vt[cur][16 + cN][8 * g],
                            *(const v8h*)&Vt[cur][16 + cN][16 + 8 * g]);
      acc1 = WMMA_F16(av1, pB0, acc1);
      const v16h av2 = cat8(*(const v8h*)&Vt[cur][32 + cN][8 * g],
                            *(const v8h*)&Vt[cur][32 + cN][16 + 8 * g]);
      acc2 = WMMA_F16(av2, pB0, acc2);
      const v16h av3 = cat8(*(const v8h*)&Vt[cur][48 + cN][8 * g],
                            *(const v8h*)&Vt[cur][48 + cN][16 + 8 * g]);
      acc3 = WMMA_F16(av3, pB0, acc3);
    }
    if (full) {   // keys 32..63 (skipped above the diagonal)
      const v16h pB1 = cat8(*(const v8h*)&Pst[w][cN][32 + 8 * g],
                            *(const v8h*)&Pst[w][cN][48 + 8 * g]);
      const v16h bv0 = cat8(*(const v8h*)&Vt[cur][cN][32 + 8 * g],
                            *(const v8h*)&Vt[cur][cN][48 + 8 * g]);
      acc0 = WMMA_F16(bv0, pB1, acc0);
      const v16h bv1 = cat8(*(const v8h*)&Vt[cur][16 + cN][32 + 8 * g],
                            *(const v8h*)&Vt[cur][16 + cN][48 + 8 * g]);
      acc1 = WMMA_F16(bv1, pB1, acc1);
      const v16h bv2 = cat8(*(const v8h*)&Vt[cur][32 + cN][32 + 8 * g],
                            *(const v8h*)&Vt[cur][32 + cN][48 + 8 * g]);
      acc2 = WMMA_F16(bv2, pB1, acc2);
      const v16h bv3 = cat8(*(const v8h*)&Vt[cur][48 + cN][32 + 8 * g],
                            *(const v8h*)&Vt[cur][48 + cN][48 + 8 * g]);
      acc3 = WMMA_F16(bv3, pB1, acc3);
    }

    // publish next buffer: wave 0 drains TDM, everyone syncs
    if (wu == 0) __builtin_amdgcn_s_wait_tensorcnt(0);
    __syncthreads();
  }

  // ---- epilogue: d contiguous over v -> packed b128 stores ----
  {
    const float rl = 1.0f / lrow;
    float* orow = Oh + (size_t)rowG * DHEAD;
    float4 o;
    o.x = acc0[0] * rl; o.y = acc0[1] * rl; o.z = acc0[2] * rl; o.w = acc0[3] * rl;
    *(float4*)(orow + 8 * g)          = o;
    o.x = acc0[4] * rl; o.y = acc0[5] * rl; o.z = acc0[6] * rl; o.w = acc0[7] * rl;
    *(float4*)(orow + 8 * g + 4)      = o;
    o.x = acc1[0] * rl; o.y = acc1[1] * rl; o.z = acc1[2] * rl; o.w = acc1[3] * rl;
    *(float4*)(orow + 16 + 8 * g)     = o;
    o.x = acc1[4] * rl; o.y = acc1[5] * rl; o.z = acc1[6] * rl; o.w = acc1[7] * rl;
    *(float4*)(orow + 16 + 8 * g + 4) = o;
    o.x = acc2[0] * rl; o.y = acc2[1] * rl; o.z = acc2[2] * rl; o.w = acc2[3] * rl;
    *(float4*)(orow + 32 + 8 * g)     = o;
    o.x = acc2[4] * rl; o.y = acc2[5] * rl; o.z = acc2[6] * rl; o.w = acc2[7] * rl;
    *(float4*)(orow + 32 + 8 * g + 4) = o;
    o.x = acc3[0] * rl; o.y = acc3[1] * rl; o.z = acc3[2] * rl; o.w = acc3[3] * rl;
    *(float4*)(orow + 48 + 8 * g)     = o;
    o.x = acc3[4] * rl; o.y = acc3[5] * rl; o.z = acc3[6] * rl; o.w = acc3[7] * rl;
    *(float4*)(orow + 48 + 8 * g + 4) = o;
  }
}

extern "C" void kernel_launch(void* const* d_in, const int* in_sizes, int n_in,
                              void* d_out, int out_size, void* d_ws, size_t ws_size,
                              hipStream_t stream) {
  const float* Q = (const float*)d_in[0];
  const float* K = (const float*)d_in[1];
  const float* V = (const float*)d_in[2];
  // d_in[3] is the causal mask; causality is computed analytically in-kernel.
  float* O = (float*)d_out;

  const size_t elems = (size_t)NHEAD * S_LEN * DHEAD;      // 4M elements
  _Float16* K16  = (_Float16*)d_ws;                        // 8 MB
  _Float16* V16t = (_Float16*)d_ws + elems;                // 8 MB

  conv_k<<<dim3((unsigned)(elems / (4 * 256))), 256, 0, stream>>>(K, K16);
  conv_v<<<dim3((unsigned)(elems / (16 * 256))), 256, 0, stream>>>(V, V16t);
  fa_fwd_causal<<<dim3(S_LEN / BLK_M, NHEAD), 128, 0, stream>>>(Q, K16, V16t, O);
}
